// LinRecAttention_67181878444214
// MI455X (gfx1250) — matrix operand order repaired and checked by reading, not compiled, not run on hardware
//
#include <hip/hip_runtime.h>
#include <hip/hip_bf16.h>

// LinRecAttention for MI455X (gfx1250, wave32, WMMA + async-to-LDS)
// B=4, N=1024, D=512, H=8, HD=64

#define BB 4
#define NN 1024
#define DDIM 512
#define HH 8
#define HDIM 64
#define BN (BB * NN)   // 4096

typedef __attribute__((ext_vector_type(16))) __bf16 v16bf;
typedef __attribute__((ext_vector_type(8)))  __bf16 v8bf;
typedef __attribute__((ext_vector_type(8)))  float  v8f;
typedef __attribute__((ext_vector_type(2)))  float  v2f;
typedef __attribute__((ext_vector_type(4)))  int    v4i;

#define GAS __attribute__((address_space(1)))
#define LAS __attribute__((address_space(3)))

#if __has_builtin(__builtin_amdgcn_global_load_async_to_lds_b128) && \
    __has_builtin(__builtin_amdgcn_s_wait_asynccnt)
#define HAVE_ASYNC 1
#else
#define HAVE_ASYNC 0
#endif

// ---------------------------------------------------------------------------
// Kernel 0: one-time f32 -> bf16 conversion (8 elems / thread)
// ---------------------------------------------------------------------------
__global__ void cvt_bf16_kernel(const float* __restrict__ src,
                                __bf16* __restrict__ dst) {
  const int i = (blockIdx.x * 256 + threadIdx.x) * 8;
  float4 a = *(const float4*)(src + i);
  float4 b = *(const float4*)(src + i + 4);
  v8bf o;
  o[0] = (__bf16)a.x; o[1] = (__bf16)a.y; o[2] = (__bf16)a.z; o[3] = (__bf16)a.w;
  o[4] = (__bf16)b.x; o[5] = (__bf16)b.y; o[6] = (__bf16)b.z; o[7] = (__bf16)b.w;
  *(v8bf*)(dst + i) = o;
}

// ---------------------------------------------------------------------------
// bf16 16x16 tile GEMM over K=512: C = Xtile(16xK) * Wtile(16xK)^T, bf16 in,
// f32 accumulate.  Inner step: 4x global_load_b128 + 1x v_wmma (no cvt).
// A-frag (16-bit A 16x32): lane m=l16; elem i -> K=kk+16*hsel+i.
// B-frag (32x16): lane n=l16, elem i ->
// K=kk+16*hsel+i.
// ---------------------------------------------------------------------------
__device__ __forceinline__ v8f gemm_tile_bf16(const __bf16* __restrict__ xrow,
                                              const __bf16* __restrict__ wrow,
                                              int hsel) {
  v8f c = {};
#pragma unroll 4
  for (int kk = 0; kk < DDIM; kk += 32) {
    const __bf16* pa = xrow + kk + hsel * 16;
    v8bf alo = *(const v8bf*)pa;
    v8bf ahi = *(const v8bf*)(pa + 8);
    v16bf a;
#pragma unroll
    for (int i = 0; i < 8; ++i) { a[i] = alo[i]; a[8 + i] = ahi[i]; }
    v16bf b = *(const v16bf*)(wrow + kk + hsel * 16);
    c = __builtin_amdgcn_wmma_f32_16x16x32_bf16(false, a, false, b,
                                                (short)0, c, false, false);
  }
  return c;
}

// ---------------------------------------------------------------------------
// Kernel 1: q/k/v = x @ W^T + b  (+elu for q,k), scattered to (b,h,n,hd)
// grid: (BN/16, D/16, 3), block: 32 (one wave per 16x16 output tile)
// ---------------------------------------------------------------------------
__global__ void qkv_proj_kernel(const __bf16* __restrict__ x,
                                const __bf16* __restrict__ Wq,
                                const __bf16* __restrict__ Wk,
                                const __bf16* __restrict__ Wv,
                                const float* __restrict__ bq,
                                const float* __restrict__ bk,
                                const float* __restrict__ bv,
                                float* __restrict__ qo,
                                float* __restrict__ ko,
                                float* __restrict__ vo) {
  const int lane = threadIdx.x;
  const int l16 = lane & 15, hsel = lane >> 4;
  const int mt = blockIdx.x, nt = blockIdx.y, which = blockIdx.z;

  const __bf16* W   = (which == 0) ? Wq : (which == 1) ? Wk : Wv;
  const float* bias = (which == 0) ? bq : (which == 1) ? bk : bv;
  float*       outp = (which == 0) ? qo : (which == 1) ? ko : vo;
  const bool   do_elu = (which < 2);

  v8f c = gemm_tile_bf16(x + (size_t)(mt * 16 + l16) * DDIM,
                         W + (size_t)(nt * 16 + l16) * DDIM, hsel);

  const int ncol = nt * 16 + l16;
  const float bval = bias[ncol];
  const int h = ncol >> 6, hd = ncol & 63;

#pragma unroll
  for (int g = 0; g < 8; ++g) {
    int row  = mt * 16 + g + hsel * 8;   // 0..BN-1
    int bidx = row >> 10;
    int n    = row & (NN - 1);
    float val = c[g] + bval;
    if (do_elu) val = (val > 0.f) ? val : expm1f(val);   // elu(alpha=1)
    outp[(((size_t)bidx * HH + h) * NN + n) * HDIM + hd] = val;
  }
}

// ---------------------------------------------------------------------------
// Kernel 2: in-place L2 normalize rows of 64, x / max(||x||, 1e-12)
// grid: (B*H*N/8, 2), block 256 (one row per wave; y picks q vs k)
// ---------------------------------------------------------------------------
__global__ void l2norm_kernel(float* __restrict__ q, float* __restrict__ k) {
  const int lane = threadIdx.x & 31, wave = threadIdx.x >> 5;
  const size_t row = (size_t)blockIdx.x * 8 + wave;
  float* base = ((blockIdx.y == 0) ? q : k) + row * HDIM + lane * 2;
  float a = base[0], b = base[1];
  float s = a * a + b * b;
#pragma unroll
  for (int off = 16; off >= 1; off >>= 1) s += __shfl_xor(s, off, 32);
  float scale = 1.f / fmaxf(sqrtf(s), 1e-12f);
  base[0] = a * scale;
  base[1] = b * scale;
}

// ---------------------------------------------------------------------------
// Async (or fallback) copy of one 16x64 f32 chunk (4 KB, contiguous) to LDS.
// 128 threads * 16 B * 2 passes per matrix.
// Builtin prototype (probed via round-2 diagnostic):
//   void __builtin_amdgcn_global_load_async_to_lds_b128(
//       v4i __attribute__((address_space(1)))*,   // global src
//       v4i __attribute__((address_space(3)))*,   // LDS dst
//       int imm_offset, int imm_cpol);
// ---------------------------------------------------------------------------
__device__ __forceinline__ void chunk_prefetch(const float* __restrict__ gq,
                                               const float* __restrict__ gk,
                                               const float* __restrict__ gv,
                                               float* lq, float* lk, float* lv,
                                               int tid) {
#pragma unroll
  for (int p = 0; p < 2; ++p) {
    const int off = (p * 128 + tid) * 4;
#if HAVE_ASYNC
    __builtin_amdgcn_global_load_async_to_lds_b128(
        (GAS v4i*)(gq + off), (LAS v4i*)(lq + off), 0, 0);
    __builtin_amdgcn_global_load_async_to_lds_b128(
        (GAS v4i*)(gk + off), (LAS v4i*)(lk + off), 0, 0);
    __builtin_amdgcn_global_load_async_to_lds_b128(
        (GAS v4i*)(gv + off), (LAS v4i*)(lv + off), 0, 0);
#else
    *(float4*)(lq + off) = *(const float4*)(gq + off);
    *(float4*)(lk + off) = *(const float4*)(gk + off);
    *(float4*)(lv + off) = *(const float4*)(gv + off);
#endif
  }
}

__device__ __forceinline__ void wait_async_all() {
#if HAVE_ASYNC
  __builtin_amdgcn_s_wait_asynccnt(0);
#endif
}

// ---------------------------------------------------------------------------
// Kernel 3: chunked causal linear attention, fp32 via V_WMMA_F32_16X16X4,
// double-buffered async Q/K/V staging in LDS.
// One block per (b,h); 4 waves, wave w owns e-columns [16w,16w+16) of the
// 64x64 state S (LDS).  Chunk T=16: out = Q@S_prev + mask(Q K^T)@V ; S += K^T@V
// f32 frags: A 16x4: lane m=l16, K=2*hsel+{0,1};  B 4x16: lane n=l16,
// K=2*hsel+{0,1};  C: VGPR g -> M=g+8*hsel, N=l16.
// ---------------------------------------------------------------------------
__global__ void attn_kernel(const float* __restrict__ q,
                            const float* __restrict__ k,
                            const float* __restrict__ v,
                            float* __restrict__ out) {
  __shared__ __align__(16) float S[HDIM][HDIM];    // 16 KB state [d][e]
  __shared__ __align__(16) float Ash[16][16];      // masked Q K^T tile
  __shared__ __align__(16) float Qs[2][16][HDIM];  // double-buffered chunks
  __shared__ __align__(16) float Ks[2][16][HDIM];
  __shared__ __align__(16) float Vs[2][16][HDIM];

  const int tid = threadIdx.x, lane = tid & 31, wave = tid >> 5;
  const int l16 = lane & 15, hsel = lane >> 4, khalf = 2 * hsel;
  const int bh = blockIdx.x, b = bh >> 3, h = bh & 7;
  const float* qb = q + (size_t)bh * NN * HDIM;
  const float* kb = k + (size_t)bh * NN * HDIM;
  const float* vb = v + (size_t)bh * NN * HDIM;
  const int ew = wave * 16 + l16;                  // this wave's e column

  for (int i = tid; i < HDIM * HDIM; i += 128) ((float*)S)[i] = 0.f;

  chunk_prefetch(qb, kb, vb, &Qs[0][0][0], &Ks[0][0][0], &Vs[0][0][0], tid);

  for (int ci = 0; ci < NN / 16; ++ci) {
    const int cur = ci & 1;
    wait_async_all();
    __syncthreads();   // staging[cur] ready; all waves done with staging[cur^1]

    if (ci + 1 < NN / 16) {
      const int n1 = (ci + 1) * 16;
      chunk_prefetch(qb + (size_t)n1 * HDIM, kb + (size_t)n1 * HDIM,
                     vb + (size_t)n1 * HDIM,
                     &Qs[cur ^ 1][0][0], &Ks[cur ^ 1][0][0], &Vs[cur ^ 1][0][0],
                     tid);
    }
    const float (*Qc)[HDIM] = Qs[cur];
    const float (*Kc)[HDIM] = Ks[cur];
    const float (*Vc)[HDIM] = Vs[cur];

    // Q A-frags: qa[f] covers contraction d = 4f+khalf+{0,1}, row m=l16
    v2f qa[16];
#pragma unroll
    for (int f = 0; f < 16; ++f)
      qa[f] = *(const v2f*)&Qc[l16][4 * f + khalf];

    // V B-frags: vf_[f] covers contraction token t = 4f+khalf+{0,1}, col ew
    v2f vf_[4];
#pragma unroll
    for (int f = 0; f < 4; ++f) {
      int t = 4 * f + khalf;
      vf_[f][0] = Vc[t][ew];
      vf_[f][1] = Vc[t + 1][ew];
    }

    // inter-chunk: c = Q @ S_prev (e-slice), contraction over d=64
    v8f c = {};
#pragma unroll
    for (int f = 0; f < 16; ++f) {
      int d = 4 * f + khalf;
      v2f sb; sb[0] = S[d][ew]; sb[1] = S[d + 1][ew];
      c = __builtin_amdgcn_wmma_f32_16x16x4_f32(false, qa[f], false, sb,
                                                (short)0, c, false, false);
    }

    // intra-chunk attention tile (wave 0): Ash = mask(Q K^T)
    if (wave == 0) {
      v8f ac = {};
#pragma unroll
      for (int f = 0; f < 16; ++f) {
        int d = 4 * f + khalf;
        v2f kf2 = *(const v2f*)&Kc[l16][d];  // key row j=l16
        ac = __builtin_amdgcn_wmma_f32_16x16x4_f32(false, qa[f], false, kf2,
                                                   (short)0, ac, false, false);
      }
#pragma unroll
      for (int g = 0; g < 8; ++g) {
        int i = g + hsel * 8, j = l16;
        Ash[i][j] = (j <= i) ? ac[g] : 0.f;  // causal (inclusive) mask
      }
    }
    __syncthreads();

    // c += A_masked @ V (contraction over 16 tokens)
#pragma unroll
    for (int f = 0; f < 4; ++f) {
      int t = 4 * f + khalf;
      v2f af = *(const v2f*)&Ash[l16][t];
      c = __builtin_amdgcn_wmma_f32_16x16x4_f32(false, af, false, vf_[f],
                                                (short)0, c, false, false);
    }

    // store chunk output into (b, n, h*HD + e) layout
    const int n0 = ci * 16;
#pragma unroll
    for (int g = 0; g < 8; ++g) {
      int n = n0 + g + hsel * 8;
      out[((size_t)b * NN + n) * DDIM + h * HDIM + ew] = c[g];
    }

    // state update: S += K^T @ V (4 d-tiles; wave-private e columns)
#pragma unroll
    for (int dt = 0; dt < 4; ++dt) {
      v8f cs = {};
      const int d = dt * 16 + l16;
#pragma unroll
      for (int f = 0; f < 4; ++f) {
        int t = 4 * f + khalf;
        v2f kf2; kf2[0] = Kc[t][d]; kf2[1] = Kc[t + 1][d];
        cs = __builtin_amdgcn_wmma_f32_16x16x4_f32(false, kf2, false, vf_[f],
                                                   (short)0, cs, false, false);
      }
#pragma unroll
      for (int g = 0; g < 8; ++g) {
        int dd = dt * 16 + g + hsel * 8;
        S[dd][ew] += cs[g];
      }
    }
  }
}

// ---------------------------------------------------------------------------
// Kernel 4: out = attn @ Wo^T + bo, plain (B*N, D) row-major f32 output
// grid: (BN/16, D/16), block 32
// ---------------------------------------------------------------------------
__global__ void out_proj_kernel(const __bf16* __restrict__ X,
                                const __bf16* __restrict__ W,
                                const float* __restrict__ bias,
                                float* __restrict__ out) {
  const int lane = threadIdx.x;
  const int l16 = lane & 15, hsel = lane >> 4;
  const int mt = blockIdx.x, nt = blockIdx.y;

  v8f c = gemm_tile_bf16(X + (size_t)(mt * 16 + l16) * DDIM,
                         W + (size_t)(nt * 16 + l16) * DDIM, hsel);

  const int ncol = nt * 16 + l16;
  const float bval = bias[ncol];
#pragma unroll
  for (int g = 0; g < 8; ++g) {
    int row = mt * 16 + g + hsel * 8;
    out[(size_t)row * DDIM + ncol] = c[g] + bval;
  }
}

// ---------------------------------------------------------------------------
extern "C" void kernel_launch(void* const* d_in, const int* in_sizes, int n_in,
                              void* d_out, int out_size, void* d_ws, size_t ws_size,
                              hipStream_t stream) {
  const float* x  = (const float*)d_in[0];
  const float* Wq = (const float*)d_in[1];
  const float* Wk = (const float*)d_in[2];
  const float* Wv = (const float*)d_in[3];
  const float* Wo = (const float*)d_in[4];
  const float* bq = (const float*)d_in[5];
  const float* bk = (const float*)d_in[6];
  const float* bv = (const float*)d_in[7];
  const float* bo = (const float*)d_in[8];
  float* out = (float*)d_out;

  // workspace layout (42 MB total):
  //   q,k,v (B,H,N,HD) f32 + attn out (B,N,D) f32     = 32 MB
  //   bf16 copies of x, Wq..Wo, attn out               = 10 MB
  const size_t NE = (size_t)BN * DDIM;   // 2,097,152 elements
  const size_t WE = (size_t)DDIM * DDIM; //   262,144 elements
  float* q_ws = (float*)d_ws;
  float* k_ws = q_ws + NE;
  float* v_ws = k_ws + NE;
  float* a_ws = v_ws + NE;
  __bf16* x_bf  = (__bf16*)(a_ws + NE);
  __bf16* wq_bf = x_bf + NE;
  __bf16* wk_bf = wq_bf + WE;
  __bf16* wv_bf = wk_bf + WE;
  __bf16* wo_bf = wv_bf + WE;
  __bf16* a_bf  = wo_bf + WE;

  // 0) one-time bf16 conversions (2048 elems/block)
  cvt_bf16_kernel<<<NE / 2048, 256, 0, stream>>>(x, x_bf);
  cvt_bf16_kernel<<<WE / 2048, 256, 0, stream>>>(Wq, wq_bf);
  cvt_bf16_kernel<<<WE / 2048, 256, 0, stream>>>(Wk, wk_bf);
  cvt_bf16_kernel<<<WE / 2048, 256, 0, stream>>>(Wv, wv_bf);
  cvt_bf16_kernel<<<WE / 2048, 256, 0, stream>>>(Wo, wo_bf);

  // 1) q/k/v projections (bf16 WMMA, f32 accumulate) + elu
  dim3 g1(BN / 16, DDIM / 16, 3);
  qkv_proj_kernel<<<g1, 32, 0, stream>>>(x_bf, wq_bf, wk_bf, wv_bf,
                                         bq, bk, bv, q_ws, k_ws, v_ws);

  // 2) per-head L2 normalize q and k
  dim3 g2((BB * HH * NN) / 8, 2);
  l2norm_kernel<<<g2, 256, 0, stream>>>(q_ws, k_ws);

  // 3) causal linear-attention scan (fp32 WMMA, async double-buffered LDS)
  attn_kernel<<<BB * HH, 128, 0, stream>>>(q_ws, k_ws, v_ws, a_ws);

  // 4) output projection
  cvt_bf16_kernel<<<NE / 2048, 256, 0, stream>>>(a_ws, a_bf);
  dim3 g4(BN / 16, DDIM / 16);
  out_proj_kernel<<<g4, 32, 0, stream>>>(a_bf, wo_bf, bo, out);
}
